// GCN_24232205484380
// MI455X (gfx1250) — matrix operand order repaired and checked by reading, not compiled, not run on hardware
//
#include <hip/hip_runtime.h>
#include <hip/hip_fp16.h>
#include <math.h>

typedef __attribute__((ext_vector_type(16))) _Float16 v16h;
typedef __attribute__((ext_vector_type(8)))  float    v8f;

#define THREADS 256

// ---------------------------------------------------------------------------
// Degree / norm precompute
// ---------------------------------------------------------------------------
__global__ void k_init_deg(float* __restrict__ deg, int N) {
    int i = blockIdx.x * blockDim.x + threadIdx.x;
    if (i < N) deg[i] = 1.0f;                 // self-loop contributes 1
}

__global__ void k_degree(const int* __restrict__ col, float* __restrict__ deg, int E) {
    int e = blockIdx.x * blockDim.x + threadIdx.x;
    if (e < E) atomicAdd(&deg[col[e]], 1.0f);
}

__global__ void k_dis(const float* __restrict__ deg, float* __restrict__ dis, int N) {
    int i = blockIdx.x * blockDim.x + threadIdx.x;
    if (i < N) dis[i] = rsqrtf(deg[i]);       // deg >= 1 always (self-loops)
}

// ---------------------------------------------------------------------------
// Layer-1 aggregation of RAW 4-dim features:  agg1 = A_norm @ x
// (aggregate-then-transform: 16x less scatter traffic than 64-dim messages)
// ---------------------------------------------------------------------------
__global__ void k_init_agg1(const float* __restrict__ x, const float* __restrict__ dis,
                            float* __restrict__ agg1, int N) {
    int idx = blockIdx.x * blockDim.x + threadIdx.x;
    if (idx < N * 4) {
        int i = idx >> 2;
        float d = dis[i];
        agg1[idx] = x[idx] * d * d;           // self-loop term
    }
}

__global__ void k_norm_agg1(const int* __restrict__ row, const int* __restrict__ col,
                            const float* __restrict__ dis, const float4* __restrict__ x4,
                            float* __restrict__ nrm, float* __restrict__ agg1, int E) {
    int e = blockIdx.x * blockDim.x + threadIdx.x;
    if (e >= E) return;
    int r = row[e], c = col[e];
    float nv = dis[r] * dis[c];
    nrm[e] = nv;                              // cache norm for the layer-2 pass
    float4 xv = x4[r];
    atomicAdd(&agg1[c * 4 + 0], xv.x * nv);
    atomicAdd(&agg1[c * 4 + 1], xv.y * nv);
    atomicAdd(&agg1[c * 4 + 2], xv.z * nv);
    atomicAdd(&agg1[c * 4 + 3], xv.w * nv);
}

// ---------------------------------------------------------------------------
// Fused dense transforms via WMMA:
//   h1 = relu(agg1 @ W1 + b1)   [16x4 @ 4x64, K padded into 16x16x32 f16]
//   p2 = h1 @ W2                [16x64 @ 64x32, 2x2 chained WMMAs]
// One wave per 16-node tile; h1 staged in LDS (f16) between the two GEMMs.
// ---------------------------------------------------------------------------
__global__ __launch_bounds__(256) void k_gemm_fused(
    const float* __restrict__ agg1,
    const float* __restrict__ W1, const float* __restrict__ b1,
    const float* __restrict__ W2,
    float* __restrict__ p2, int nTiles, int N)
{
    __shared__ _Float16 h1s[8][16][64];       // 8 waves * 2KB = 16KB
    const int lane = threadIdx.x & 31;
    const int wid  = threadIdx.x >> 5;
    const int m    = lane & 15;
    const int hi   = (lane >= 16) ? 1 : 0;

    int tile = blockIdx.x * 8 + wid;
    if (tile >= nTiles) tile = nTiles - 1;    // clamp: redundant identical work, no divergence at barriers/WMMA
    const int row0 = tile * 16;

    // ---- A fragment for layer 1 (16x32 f16 layout; only K=0..3 nonzero) ----
    // 16-bit A layout: lanes 0-15 halves 0-7 -> K 0..7 (we need K 0..3)
    v16h a1 = {};
    if (lane < 16) {
        int rr = row0 + m; if (rr >= N) rr = N - 1;
        const float* xr = agg1 + (size_t)rr * 4;
        a1[0] = (_Float16)xr[0];
        a1[1] = (_Float16)xr[1];
        a1[2] = (_Float16)xr[2];
        a1[3] = (_Float16)xr[3];
    }

    // ---- Layer 1: 4 col-tiles of 16 over the 64 outputs ----
    for (int t = 0; t < 4; ++t) {
        // B fragment (32x16 f16), contiguous k order chosen here: lane -> N=lane%16; half i -> K = 16*hi + i.
        // Only K=0..3 valid -> lanes 0-15, halves 0..3.
        v16h b = {};
        if (lane < 16) {
            int n = t * 16 + lane;
            b[0] = (_Float16)W1[0 * 64 + n];
            b[1] = (_Float16)W1[1 * 64 + n];
            b[2] = (_Float16)W1[2 * 64 + n];
            b[3] = (_Float16)W1[3 * 64 + n];
        }
        v8f c = {};
        c = __builtin_amdgcn_wmma_f32_16x16x32_f16(false, a1, false, b,
                                                   (short)0, c, false, false);
        // C layout: VGPR j -> M = 8*hi + j, N = lane%16 (+16*t)
        int n = t * 16 + m;
        float bias = b1[n];
#pragma unroll
        for (int j = 0; j < 8; ++j) {
            float v = c[j] + bias;
            h1s[wid][hi * 8 + j][n] = (_Float16)(v > 0.0f ? v : 0.0f);  // ReLU
        }
    }
    __syncthreads();   // all 256 threads reach this (tiles clamped, no divergence)

    // ---- Re-pack h1 from LDS into two A fragments (K 0..31 and K 32..63) ----
    v16h a2lo = {}, a2hi = {};
    {
        const int kb = hi * 16;
#pragma unroll
        for (int i = 0; i < 16; ++i) {
            a2lo[i] = h1s[wid][m][kb + i];
            a2hi[i] = h1s[wid][m][32 + kb + i];
        }
    }

    // ---- Layer 2: 2 col-tiles x 2 K-blocks, chained accumulation ----
#pragma unroll
    for (int t = 0; t < 2; ++t) {
        int n = t * 16 + m;
        const int kb = hi * 16;
        v16h bA = {}, bB = {};
#pragma unroll
        for (int i = 0; i < 16; ++i) {
            bA[i] = (_Float16)W2[(kb + i) * 32 + n];        // K block 0..31
            bB[i] = (_Float16)W2[(32 + kb + i) * 32 + n];   // K block 32..63
        }
        v8f c = {};
        c = __builtin_amdgcn_wmma_f32_16x16x32_f16(false, a2lo, false, bA,
                                                   (short)0, c, false, false);
        c = __builtin_amdgcn_wmma_f32_16x16x32_f16(false, a2hi, false, bB,
                                                   (short)0, c, false, false);
#pragma unroll
        for (int j = 0; j < 8; ++j) {
            int rr = row0 + hi * 8 + j;
            if (rr < N) p2[(size_t)rr * 32 + n] = c[j];
        }
    }
}

// ---------------------------------------------------------------------------
// Layer-2 aggregation of 32-dim transformed features: agg2 = A_norm @ p2
// One wave per edge: lane = feature -> coalesced 128B gather + 32 f32 atomics.
// ---------------------------------------------------------------------------
__global__ void k_init_agg2(const float* __restrict__ p2, const float* __restrict__ dis,
                            float* __restrict__ agg2, int N) {
    int idx = blockIdx.x * blockDim.x + threadIdx.x;
    if (idx < N * 32) {
        int i = idx >> 5;
        float d = dis[i];
        agg2[idx] = p2[idx] * d * d;          // self-loop term
    }
}

__global__ void k_agg2(const int* __restrict__ row, const int* __restrict__ col,
                       const float* __restrict__ nrm, const float* __restrict__ p2,
                       float* __restrict__ agg2, long long total) {
    long long idx = (long long)blockIdx.x * blockDim.x + threadIdx.x;
    if (idx >= total) return;
    int e = (int)(idx >> 5);
    int f = (int)(idx & 31);
    int r = row[e], c = col[e];
    atomicAdd(&agg2[(size_t)c * 32 + f], p2[(size_t)r * 32 + f] * nrm[e]);
}

// ---------------------------------------------------------------------------
// Pool + FC head
// ---------------------------------------------------------------------------
__global__ void k_zero32(float* __restrict__ g) { g[threadIdx.x] = 0.0f; }

__global__ __launch_bounds__(256) void k_pool(const float* __restrict__ agg2,
                                              const float* __restrict__ b2,
                                              float* __restrict__ gpart, int N) {
    __shared__ float s[256];
    int f   = threadIdx.x & 31;
    int sub = threadIdx.x >> 5;               // node sub-slot within block (0..7)
    float bias = b2[f];
    float acc = 0.0f;
    for (int i = blockIdx.x * 8 + sub; i < N; i += gridDim.x * 8) {
        float v = agg2[(size_t)i * 32 + f] + bias;
        acc += (v > 0.0f ? v : 0.0f);         // ReLU before mean
    }
    s[threadIdx.x] = acc;
    __syncthreads();
    if (threadIdx.x < 32) {
        float t = 0.0f;
#pragma unroll
        for (int w = 0; w < 8; ++w) t += s[w * 32 + f];
        atomicAdd(&gpart[f], t);
    }
}

__global__ void k_final(const float* __restrict__ gpart, const float* __restrict__ Wfc,
                        const float* __restrict__ bfc, float* __restrict__ out, int N) {
    int f = threadIdx.x;                      // 32 threads, one wave
    float v = (gpart[f] / (float)N) * Wfc[f];
#pragma unroll
    for (int off = 16; off > 0; off >>= 1) v += __shfl_down(v, off, 32);
    if (f == 0) out[0] = 1.0f / (1.0f + expf(-(v + bfc[0])));
}

// ---------------------------------------------------------------------------
// Launch
// ---------------------------------------------------------------------------
extern "C" void kernel_launch(void* const* d_in, const int* in_sizes, int n_in,
                              void* d_out, int out_size, void* d_ws, size_t ws_size,
                              hipStream_t stream) {
    const float* x   = (const float*)d_in[0];
    const int*   ei  = (const int*)d_in[1];   // edge_index [2, E]
    const float* W1  = (const float*)d_in[2];
    const float* b1  = (const float*)d_in[3];
    const float* W2  = (const float*)d_in[4];
    const float* b2  = (const float*)d_in[5];
    const float* Wfc = (const float*)d_in[6];
    const float* bfc = (const float*)d_in[7];
    float* out = (float*)d_out;

    const int N = in_sizes[0] / 4;
    const int E = in_sizes[1] / 2;
    const int* row = ei;
    const int* col = ei + E;

    // Workspace carve-out (~66 MB): deg, dis, per-edge norm, agg1, p2, agg2, gpart
    char* ws = (char*)d_ws;
    size_t off = 0;
    auto wsAlloc = [&](size_t bytes) -> void* {
        void* p = (void*)(ws + off);
        off += (bytes + 255) & ~(size_t)255;
        return p;
    };
    float* deg   = (float*)wsAlloc((size_t)N * 4);
    float* dis   = (float*)wsAlloc((size_t)N * 4);
    float* nrm   = (float*)wsAlloc((size_t)E * 4);
    float* agg1  = (float*)wsAlloc((size_t)N * 4 * 4);
    float* p2    = (float*)wsAlloc((size_t)N * 32 * 4);
    float* agg2  = (float*)wsAlloc((size_t)N * 32 * 4);
    float* gpart = (float*)wsAlloc(32 * 4);
    (void)ws_size; (void)n_in; (void)out_size;

    const int gN = (N + THREADS - 1) / THREADS;
    const int gE = (E + THREADS - 1) / THREADS;

    k_init_deg <<<gN, THREADS, 0, stream>>>(deg, N);
    k_degree   <<<gE, THREADS, 0, stream>>>(col, deg, E);
    k_dis      <<<gN, THREADS, 0, stream>>>(deg, dis, N);

    k_init_agg1<<<(N * 4 + THREADS - 1) / THREADS, THREADS, 0, stream>>>(x, dis, agg1, N);
    k_norm_agg1<<<gE, THREADS, 0, stream>>>(row, col, dis, (const float4*)x, nrm, agg1, E);

    const int nTiles = (N + 15) / 16;
    k_gemm_fused<<<(nTiles + 7) / 8, THREADS, 0, stream>>>(agg1, W1, b1, W2, p2, nTiles, N);

    k_init_agg2<<<(N * 32 + THREADS - 1) / THREADS, THREADS, 0, stream>>>(p2, dis, agg2, N);
    const long long tot = (long long)E * 32;
    k_agg2<<<(int)((tot + THREADS - 1) / THREADS), THREADS, 0, stream>>>(row, col, nrm, p2, agg2, tot);

    k_zero32<<<1, 32, 0, stream>>>(gpart);
    k_pool  <<<512, THREADS, 0, stream>>>(agg2, b2, gpart, N);
    k_final <<<1, 32, 0, stream>>>(gpart, Wfc, bfc, out, N);
}
